// Mamba2ResidualBlock_26113401160185
// MI455X (gfx1250) — matrix operand order repaired and checked by reading, not compiled, not run on hardware
//
#include <hip/hip_runtime.h>
#include <cstddef>

#define B_      2
#define L_      2048
#define DMODEL  1024
#define DSTATE  128
#define DCONV   4
#define NHEADS  32
#define HEADDIM 64
#define DINNER  2048
#define CONVDIM 2304
#define DIN_    4384
#define CHUNK   256
#define NCHUNK  8
#define NTOK    (B_*L_)
#define EPSV    1e-5f

typedef __bf16 bf16;
typedef __attribute__((ext_vector_type(16))) __bf16 v16bf;
typedef __attribute__((ext_vector_type(8)))  __bf16 v8bf;
typedef __attribute__((ext_vector_type(8)))  float  v8f;

__device__ __forceinline__ float siluf(float x) { return x / (1.f + expf(-x)); }

// ---------------------------------------------------------------- utilities
__global__ void f32_to_bf16_kernel(const float* __restrict__ a, bf16* __restrict__ o, int n) {
  int i = blockIdx.x * blockDim.x + threadIdx.x;
  if (i < n) o[i] = (bf16)a[i];
}

// RMSNorm over DMODEL, emit bf16 activations for the in_proj WMMA GEMM.
__global__ void rmsnorm_in_kernel(const float* __restrict__ x, const float* __restrict__ w,
                                  bf16* __restrict__ out) {
  int t = blockIdx.x;                  // token row
  int tid = threadIdx.x;               // 256 threads, 4 elems each
  const float* xr = x + (size_t)t * DMODEL;
  float ss = 0.f;
  float xv[4];
  #pragma unroll
  for (int j = 0; j < 4; ++j) { xv[j] = xr[tid + j*256]; ss += xv[j]*xv[j]; }
  __shared__ float red[256];
  red[tid] = ss; __syncthreads();
  for (int s = 128; s > 0; s >>= 1) { if (tid < s) red[tid] += red[tid+s]; __syncthreads(); }
  float scale = rsqrtf(red[0] / (float)DMODEL + EPSV);
  #pragma unroll
  for (int j = 0; j < 4; ++j) {
    int i = tid + j*256;
    out[(size_t)t*DMODEL + i] = (bf16)(xv[j] * scale * w[i]);
  }
}

// ------------------------------------------------------- WMMA bf16 GEMM (NT)
// C[M,N] (+= addsrc) = A[M,K] * W[N,K]^T.
// Register-blocked: each wave32 owns a 64x32 macro-tile = 4 (M) x 2 (N)
// 16x16 WMMA tiles -> 8 v_wmma per k-step from 4 A-frags + 2 B-frags
// (2.7x less L2 traffic per WMMA than 1-tile-per-wave).
// A-fragment: lanes 0-15 row M=lane, 16 contiguous K at
// kb..kb+15, kb=(lane>=16)?16:0. B-fragment: lane col
// N=lane&15, 16 contiguous K at (lane>=16?16:0) -> weight rows load
// contiguously as one 32B vector.
__global__ void __launch_bounds__(256)
gemm_bf16_wmma(const bf16* __restrict__ A, const bf16* __restrict__ W,
               const float* __restrict__ addsrc, float* __restrict__ C,
               int M, int N, int K, size_t sA, size_t sW, size_t sC) {
  A += sA * blockIdx.y;  W += sW * blockIdx.y;  C += sC * blockIdx.y;
  int wave = threadIdx.x >> 5;
  int lane = threadIdx.x & 31;
  int tile = blockIdx.x * (blockDim.x >> 5) + wave;
  int MT = M >> 6, NT = N >> 5;         // 64x32 wave tiles
  if (tile >= MT * NT) return;
  int m0 = (tile / NT) << 6;
  int n0 = (tile % NT) << 5;
  const bf16* Ap = A + (size_t)(m0 + (lane & 15)) * K + ((lane >> 4) << 4);
  const bf16* Bp = W + (size_t)(n0 + (lane & 15)) * K + ((lane >> 4) << 4);
  v8f acc[4][2];
  #pragma unroll
  for (int im = 0; im < 4; ++im)
    #pragma unroll
    for (int in = 0; in < 2; ++in) acc[im][in] = (v8f){};
  for (int k = 0; k < K; k += 32) {
    v16bf av[4], bv[2];
    #pragma unroll
    for (int im = 0; im < 4; ++im) {
      const bf16* p = Ap + (size_t)(im << 4) * K + k;
      v8bf alo = *(const v8bf*)(p);
      v8bf ahi = *(const v8bf*)(p + 8);
      #pragma unroll
      for (int i = 0; i < 8; ++i) { av[im][i] = alo[i]; av[im][i+8] = ahi[i]; }
    }
    #pragma unroll
    for (int in = 0; in < 2; ++in)
      bv[in] = *(const v16bf*)(Bp + (size_t)(in << 4) * K + k);
    #pragma unroll
    for (int im = 0; im < 4; ++im)
      #pragma unroll
      for (int in = 0; in < 2; ++in)
        acc[im][in] = __builtin_amdgcn_wmma_f32_16x16x32_bf16(
            false, av[im], false, bv[in], (short)0, acc[im][in], false, false);
  }
  int rbase = ((lane >> 4) << 3);       // 0 or 8
  int cbase = lane & 15;
  #pragma unroll
  for (int im = 0; im < 4; ++im) {
    #pragma unroll
    for (int in = 0; in < 2; ++in) {
      int crow = m0 + (im << 4) + rbase;
      int ccol = n0 + (in << 4) + cbase;
      #pragma unroll
      for (int j = 0; j < 8; ++j) {
        size_t off = (size_t)(crow + j) * N + ccol;
        float v = acc[im][in][j];
        if (addsrc) v += addsrc[off];
        C[off] = v;
      }
    }
  }
}

// -------------------------------------------------------------- dt softplus
__global__ void dt_softplus_kernel(const float* __restrict__ zx, const float* __restrict__ dt_bias,
                                   const float* __restrict__ A_log,
                                   float* __restrict__ dt, float* __restrict__ dtA) {
  int idx = blockIdx.x * blockDim.x + threadIdx.x;
  if (idx >= NTOK * NHEADS) return;
  int h = idx % NHEADS, t = idx / NHEADS;
  float raw = zx[(size_t)t * DIN_ + (DIN_ - NHEADS) + h] + dt_bias[h];
  float sp = (raw > 20.f) ? raw : log1pf(expf(raw));
  dt[idx]  = sp;
  dtA[idx] = -expf(A_log[h]) * sp;
}

// -------------------------------- depthwise causal conv + SiLU + routing
__global__ void conv_silu_route_kernel(const float* __restrict__ zx,
                                       const float* __restrict__ conv_w, const float* __restrict__ conv_b,
                                       const float* __restrict__ dt, const float* __restrict__ Dvec,
                                       float* __restrict__ Xd, float* __restrict__ Yacc,
                                       float* __restrict__ Bm, float* __restrict__ Cm,
                                       bf16* __restrict__ Bmh, bf16* __restrict__ Cmh) {
  int idx = blockIdx.x * blockDim.x + threadIdx.x;
  if (idx >= NTOK * CONVDIM) return;
  int c = idx % CONVDIM;
  int t = idx / CONVDIM;
  int l = t % L_;
  float acc = conv_b[c];
  #pragma unroll
  for (int k = 0; k < DCONV; ++k) {
    int l2 = l + k - (DCONV - 1);
    if (l2 >= 0)
      acc += zx[(size_t)(t + k - (DCONV - 1)) * DIN_ + DINNER + c] * conv_w[c * DCONV + k];
  }
  float v = siluf(acc);
  if (c < DINNER) {
    int h = c >> 6;
    Xd[(size_t)t * DINNER + c]   = v * dt[t * NHEADS + h];
    Yacc[(size_t)t * DINNER + c] = Dvec[h] * v;           // seed Y with D * x
  } else if (c < DINNER + DSTATE) {
    int n = c - DINNER;
    Bm[(size_t)t * DSTATE + n] = v;  Bmh[(size_t)t * DSTATE + n] = (bf16)v;
  } else {
    int n = c - DINNER - DSTATE;
    Cm[(size_t)t * DSTATE + n] = v;  Cmh[(size_t)t * DSTATE + n] = (bf16)v;
  }
}

// -------------------------------------- per-(b,chunk,head) inclusive cumsum
__global__ void chunk_cumsum_kernel(const float* __restrict__ dtA,
                                    float* __restrict__ Acs, float* __restrict__ AcsLast) {
  int h  = blockIdx.x % NHEADS;
  int bc = blockIdx.x / NHEADS;          // b*NCHUNK + c
  int l  = threadIdx.x;
  __shared__ float s[CHUNK];
  s[l] = dtA[(size_t)(bc * CHUNK + l) * NHEADS + h];
  __syncthreads();
  for (int off = 1; off < CHUNK; off <<= 1) {
    float t = (l >= off) ? s[l - off] : 0.f;
    __syncthreads();
    s[l] += t;
    __syncthreads();
  }
  Acs[((size_t)bc * NHEADS + h) * CHUNK + l] = s[l];
  if (l == CHUNK - 1) AcsLast[bc * NHEADS + h] = s[l];
}

// ------------------------- Y_diag accumulation + per-chunk states (fp32 VALU)
__global__ void __launch_bounds__(256)
ydiag_states_kernel(const float* __restrict__ Xd, const float* __restrict__ Bm,
                    const float* __restrict__ CB, const float* __restrict__ Acs,
                    const float* __restrict__ AcsLast,
                    float* __restrict__ Yacc, float* __restrict__ states) {
  int h  = blockIdx.x % NHEADS;
  int bc = blockIdx.x / NHEADS;
  int tid = threadIdx.x;
  __shared__ bf16  xds[CHUNK][HEADDIM];   // 32 KB
  __shared__ float acs[CHUNK];
  __shared__ float dec[CHUNK];
  for (int i = tid; i < CHUNK * HEADDIM; i += 256) {
    int s = i >> 6, p = i & 63;
    xds[s][p] = (bf16)Xd[(size_t)(bc * CHUNK + s) * DINNER + h * HEADDIM + p];
  }
  float myA = Acs[((size_t)bc * NHEADS + h) * CHUNK + tid];
  acs[tid] = myA;
  float alast = AcsLast[bc * NHEADS + h];
  dec[tid] = expf(alast - myA);
  __syncthreads();
  { // phase 1: Y_diag, one output row per thread
    int l = tid;
    float acc[HEADDIM];
    #pragma unroll
    for (int p = 0; p < HEADDIM; ++p) acc[p] = 0.f;
    const float* cbrow = CB + ((size_t)bc << 16) + (size_t)l * CHUNK;
    for (int s = 0; s <= l; ++s) {
      float w = cbrow[s] * expf(myA - acs[s]);
      #pragma unroll
      for (int p = 0; p < HEADDIM; ++p) acc[p] += w * (float)xds[s][p];
    }
    float* yrow = Yacc + (size_t)(bc * CHUNK + l) * DINNER + h * HEADDIM;
    #pragma unroll
    for (int p = 0; p < HEADDIM; ++p) yrow[p] += acc[p];
  }
  __syncthreads();
  { // phase 2: states[p][n] = sum_l decay[l] * Xd[l,p] * B[l,n]
    int n  = tid & (DSTATE - 1);
    int pb = (tid >> 7) * 32;
    float st[32];
    #pragma unroll
    for (int q = 0; q < 32; ++q) st[q] = 0.f;
    for (int l2 = 0; l2 < CHUNK; ++l2) {
      float bv = Bm[(size_t)(bc * CHUNK + l2) * DSTATE + n] * dec[l2];
      #pragma unroll
      for (int q = 0; q < 32; ++q) st[q] += bv * (float)xds[l2][pb + q];
    }
    float* sp = states + ((size_t)bc * NHEADS + h) * (HEADDIM * DSTATE);
    #pragma unroll
    for (int q = 0; q < 32; ++q) sp[(size_t)(pb + q) * DSTATE + n] = st[q];
  }
}

// ---------------- sequential inter-chunk scan, in place (states -> prefixes)
__global__ void chunk_scan_kernel(const float* __restrict__ AcsLast, float* __restrict__ states) {
  int h = blockIdx.x % NHEADS;
  int b = blockIdx.x / NHEADS;
  int tid = threadIdx.x;
  float P[32];
  #pragma unroll
  for (int q = 0; q < 32; ++q) P[q] = 0.f;
  for (int c = 0; c < NCHUNK; ++c) {
    int bc = b * NCHUNK + c;
    float g = expf(AcsLast[bc * NHEADS + h]);
    float* buf = states + ((size_t)bc * NHEADS + h) * (HEADDIM * DSTATE);
    #pragma unroll
    for (int q = 0; q < 32; ++q) {
      size_t e = (size_t)tid + 256u * q;
      float sv = buf[e];
      buf[e] = P[q];              // prefix state entering this chunk
      P[q] = g * P[q] + sv;
    }
  }
}

// ---------------------------------------------- Y_off from prefix states
__global__ void __launch_bounds__(256)
yoff_kernel(const float* __restrict__ Cm, const float* __restrict__ states,
            const float* __restrict__ Acs, float* __restrict__ Yacc) {
  int h  = blockIdx.x % NHEADS;
  int bc = blockIdx.x / NHEADS;
  int tid = threadIdx.x;
  __shared__ float Ps[HEADDIM][DSTATE];   // 32 KB
  __shared__ float acs[CHUNK];
  const float* sp = states + ((size_t)bc * NHEADS + h) * (HEADDIM * DSTATE);
  for (int i = tid; i < HEADDIM * DSTATE; i += 256)
    Ps[i >> 7][i & 127] = sp[i];
  acs[tid] = Acs[((size_t)bc * NHEADS + h) * CHUNK + tid];
  __syncthreads();
  int l = tid;
  float ea = expf(acs[l]);
  float acc[HEADDIM];
  #pragma unroll
  for (int p = 0; p < HEADDIM; ++p) acc[p] = 0.f;
  const float* crow = Cm + (size_t)(bc * CHUNK + l) * DSTATE;
  for (int n = 0; n < DSTATE; ++n) {
    float cv = crow[n];
    #pragma unroll
    for (int p = 0; p < HEADDIM; ++p) acc[p] += cv * Ps[p][n];
  }
  float* yrow = Yacc + (size_t)(bc * CHUNK + l) * DINNER + h * HEADDIM;
  #pragma unroll
  for (int p = 0; p < HEADDIM; ++p) yrow[p] += ea * acc[p];
}

// ------------------------------------- gating + RMSNorm over DINNER -> bf16
__global__ void gate_rmsnorm_kernel(const float* __restrict__ Yacc, const float* __restrict__ zx,
                                    const float* __restrict__ gw, bf16* __restrict__ yn) {
  int t = blockIdx.x, tid = threadIdx.x;   // 256 threads, 8 elems each
  float yv[8];
  float ss = 0.f;
  #pragma unroll
  for (int j = 0; j < 8; ++j) {
    int i = tid + j * 256;
    float z = zx[(size_t)t * DIN_ + i];
    float y = Yacc[(size_t)t * DINNER + i] * siluf(z);
    yv[j] = y; ss += y * y;
  }
  __shared__ float red[256];
  red[tid] = ss; __syncthreads();
  for (int s = 128; s > 0; s >>= 1) { if (tid < s) red[tid] += red[tid+s]; __syncthreads(); }
  float scale = rsqrtf(red[0] / (float)DINNER + EPSV);
  #pragma unroll
  for (int j = 0; j < 8; ++j) {
    int i = tid + j * 256;
    yn[(size_t)t * DINNER + i] = (bf16)(yv[j] * scale * gw[i]);
  }
}

// ============================================================== launcher
extern "C" void kernel_launch(void* const* d_in, const int* in_sizes, int n_in,
                              void* d_out, int out_size, void* d_ws, size_t ws_size,
                              hipStream_t stream) {
  (void)in_sizes; (void)n_in; (void)out_size; (void)ws_size;
  const float* x          = (const float*)d_in[0];
  const float* norm_w     = (const float*)d_in[1];
  const float* in_proj_w  = (const float*)d_in[2];
  const float* conv_w     = (const float*)d_in[3];
  const float* conv_b     = (const float*)d_in[4];
  const float* dt_bias    = (const float*)d_in[5];
  const float* A_log      = (const float*)d_in[6];
  const float* Dvec       = (const float*)d_in[7];
  const float* gnorm_w    = (const float*)d_in[8];
  const float* out_proj_w = (const float*)d_in[9];
  float* out = (float*)d_out;

  char* ws = (char*)d_ws;
  size_t off = 0;
  auto alloc = [&](size_t bytes) -> char* {
    char* p = ws + off;
    off += (bytes + 255) & ~(size_t)255;
    return p;
  };
  bf16*  xn_bf     = (bf16*)alloc((size_t)NTOK * DMODEL * 2);
  bf16*  win_bf    = (bf16*)alloc((size_t)DIN_ * DMODEL * 2);
  bf16*  wout_bf   = (bf16*)alloc((size_t)DMODEL * DINNER * 2);
  float* zx        = (float*)alloc((size_t)NTOK * DIN_ * 4);
  float* dt        = (float*)alloc((size_t)NTOK * NHEADS * 4);
  float* dtA       = (float*)alloc((size_t)NTOK * NHEADS * 4);
  float* Bm        = (float*)alloc((size_t)NTOK * DSTATE * 4);
  float* Cm        = (float*)alloc((size_t)NTOK * DSTATE * 4);
  bf16*  Bmh       = (bf16*)alloc((size_t)NTOK * DSTATE * 2);
  bf16*  Cmh       = (bf16*)alloc((size_t)NTOK * DSTATE * 2);
  float* Xd        = (float*)alloc((size_t)NTOK * DINNER * 4);
  float* Yacc      = (float*)alloc((size_t)NTOK * DINNER * 4);
  float* Acs       = (float*)alloc((size_t)B_ * NCHUNK * NHEADS * CHUNK * 4);
  float* AcsLast   = (float*)alloc((size_t)B_ * NCHUNK * NHEADS * 4);
  float* CB        = (float*)alloc((size_t)B_ * NCHUNK * CHUNK * CHUNK * 4);
  float* states    = (float*)alloc((size_t)B_ * NCHUNK * NHEADS * HEADDIM * DSTATE * 4);
  bf16*  yn_bf     = (bf16*)alloc((size_t)NTOK * DINNER * 2);

  // weight packs
  {
    int n1 = DIN_ * DMODEL;
    f32_to_bf16_kernel<<<(n1 + 255) / 256, 256, 0, stream>>>(in_proj_w, win_bf, n1);
    int n2 = DMODEL * DINNER;
    f32_to_bf16_kernel<<<(n2 + 255) / 256, 256, 0, stream>>>(out_proj_w, wout_bf, n2);
  }
  // RMSNorm
  rmsnorm_in_kernel<<<NTOK, 256, 0, stream>>>(x, norm_w, xn_bf);
  // in_proj GEMM: [4096,1024] x [4384,1024]^T, 64x32 tile per wave
  {
    int tiles = (NTOK / 64) * (DIN_ / 32);
    gemm_bf16_wmma<<<dim3((tiles + 7) / 8, 1), 256, 0, stream>>>(
        xn_bf, win_bf, nullptr, zx, NTOK, DIN_, DMODEL, 0, 0, 0);
  }
  // dt softplus
  dt_softplus_kernel<<<(NTOK * NHEADS + 255) / 256, 256, 0, stream>>>(zx, dt_bias, A_log, dt, dtA);
  // conv + silu + routing (also seeds Yacc with D*x, builds Xd = x*dt)
  conv_silu_route_kernel<<<(NTOK * CONVDIM + 255) / 256, 256, 0, stream>>>(
      zx, conv_w, conv_b, dt, Dvec, Xd, Yacc, Bm, Cm, Bmh, Cmh);
  // per-chunk cumsum of dt*A
  chunk_cumsum_kernel<<<B_ * NCHUNK * NHEADS, CHUNK, 0, stream>>>(dtA, Acs, AcsLast);
  // CB = C * B^T per (b,chunk): batched WMMA GEMM, M=N=256, K=128
  {
    int tiles = (CHUNK / 64) * (CHUNK / 32);     // 32 waves per batch
    gemm_bf16_wmma<<<dim3((tiles + 7) / 8, B_ * NCHUNK), 256, 0, stream>>>(
        Cmh, Bmh, nullptr, CB, CHUNK, CHUNK, DSTATE,
        (size_t)CHUNK * DSTATE, (size_t)CHUNK * DSTATE, (size_t)CHUNK * CHUNK);
  }
  // Y_diag + chunk states
  ydiag_states_kernel<<<B_ * NCHUNK * NHEADS, CHUNK, 0, stream>>>(
      Xd, Bm, CB, Acs, AcsLast, Yacc, states);
  // inter-chunk scan (in place -> prefix states)
  chunk_scan_kernel<<<B_ * NHEADS, 256, 0, stream>>>(AcsLast, states);
  // Y_off
  yoff_kernel<<<B_ * NCHUNK * NHEADS, CHUNK, 0, stream>>>(Cm, states, Acs, Yacc);
  // gate + RMSNorm -> bf16
  gate_rmsnorm_kernel<<<NTOK, 256, 0, stream>>>(Yacc, zx, gnorm_w, yn_bf);
  // out_proj GEMM with fused residual: out = x + yn * Wout^T
  {
    int tiles = (NTOK / 64) * (DMODEL / 32);
    gemm_bf16_wmma<<<dim3((tiles + 7) / 8, 1), 256, 0, stream>>>(
        yn_bf, wout_bf, x, out, NTOK, DMODEL, DINNER, 0, 0, 0);
  }
}
